// SJTA_55843164782755
// MI455X (gfx1250) — matrix operand order repaired and checked by reading, not compiled, not run on hardware
//
#include <hip/hip_runtime.h>
#include <hip/hip_bf16.h>

// ---------------- CDNA5 WMMA types ----------------
typedef __attribute__((ext_vector_type(16))) __bf16 v16bf;
typedef __attribute__((ext_vector_type(8)))  float  v8f;

union BF16Frag { v16bf v; unsigned short u[16]; uint4 q[2]; };
union F32x8    { v8f   v; float f[8]; };

#define BATCH   128
#define CFEAT   1280
#define NTOK    256     // H*W
#define BD      256
#define NHEAD   4
#define HDIM    64
#define NQ      21
#define NJ      16
#define MTOT    (BATCH*NTOK)   // 32768

__device__ __forceinline__ unsigned short f2bf(float f) {
    unsigned u = __float_as_uint(f);
    u += 0x7FFFu + ((u >> 16) & 1u);    // round-to-nearest-even
    return (unsigned short)(u >> 16);
}

// ---------------- 1) pack wk|wv -> bf16 TRANSPOSED (512 cols x 1280 k, k-contiguous) ----------------
__global__ __launch_bounds__(256) void pose_pack_wkv(const float* __restrict__ wk,
                                                     const float* __restrict__ wv,
                                                     unsigned short* __restrict__ wkvT) {
    int col = blockIdx.x;                // 0..511
    const float* src = (col < 256) ? (wk + col) : (wv + (col - 256));
    for (int k = threadIdx.x; k < CFEAT; k += 256)
        wkvT[(size_t)col * CFEAT + k] = f2bf(src[(size_t)k * 256]);
}

// ---------------- 2) LayerNorm over C, write tokens bf16 m-major (32768 x 1280) ----------------
__global__ __launch_bounds__(256) void pose_ln(const float* __restrict__ img,
                                               const float* __restrict__ g,
                                               const float* __restrict__ be,
                                               unsigned short* __restrict__ tok) {
    __shared__ float rs[4][64], rs2[4][64], mean_[64], rstd_[64];
    __shared__ unsigned short tile[64][72];      // [token][c_local], stride 72 (144B)
    int b  = blockIdx.x >> 2;
    int n0 = (blockIdx.x & 3) * 64;
    int tid = threadIdx.x;
    int tn = tid & 63, tc = tid >> 6;
    const float* base = img + (size_t)b * CFEAT * NTOK + n0 + tn;
    float s = 0.f, s2 = 0.f;
    for (int c = tc; c < CFEAT; c += 4) { float x = base[(size_t)c * NTOK]; s += x; s2 += x * x; }
    rs[tc][tn] = s; rs2[tc][tn] = s2;
    __syncthreads();
    if (tc == 0) {
        float ss = rs[0][tn] + rs[1][tn] + rs[2][tn] + rs[3][tn];
        float qq = rs2[0][tn] + rs2[1][tn] + rs2[2][tn] + rs2[3][tn];
        float m = ss * (1.f / CFEAT);
        float v = qq * (1.f / CFEAT) - m * m;
        mean_[tn] = m;
        rstd_[tn] = rsqrtf(v + 1e-5f);
    }
    __syncthreads();
    float m = mean_[tn], r = rstd_[tn];
    int row = tid >> 2, seg = tid & 3;
    for (int cc = 0; cc < CFEAT; cc += 64) {
#pragma unroll
        for (int i = 0; i < 8; i++) {
            int cl = tc * 16 + 2 * i;
            int c  = cc + cl;
            float x0 = base[(size_t)c * NTOK];
            float x1 = base[(size_t)(c + 1) * NTOK];
            unsigned p = (unsigned)f2bf((x0 - m) * r * g[c] + be[c])
                       | ((unsigned)f2bf((x1 - m) * r * g[c + 1] + be[c + 1]) << 16);
            *(unsigned*)&tile[tn][cl] = p;
        }
        __syncthreads();
        uint4 q0 = *(uint4*)&tile[row][seg * 16];
        uint4 q1 = *(uint4*)&tile[row][seg * 16 + 8];
        unsigned short* dst = tok + (size_t)(b * NTOK + n0 + row) * CFEAT + cc + seg * 16;
        *(uint4*)dst = q0;
        *(uint4*)(dst + 8) = q1;
        __syncthreads();
    }
}

// ---------------- 3) fused K|V GEMM with v_wmma_f32_16x16x32_bf16 ----------------
// C(32768 x 512) = tokens(32768 x 1280, m-major) x wkvT(512 x 1280)^T + bias
// Block tile 128x128, 8 waves, each wave 2(M) x 4(N) 16x16 accumulators:
// 12 ds_load_b128 per 8 v_wmma (1.5 loads/wmma).
#define G_MT 128
#define G_NT 128
#define G_KT 32
#define T_STR 40     // LDS tile stride in ushorts (80B: 16B-aligned, conflict-free)
__global__ __launch_bounds__(256) void pose_kv_gemm(const unsigned short* __restrict__ tok,
                                                    const unsigned short* __restrict__ wkvT,
                                                    const float* __restrict__ bk,
                                                    const float* __restrict__ bv,
                                                    float* __restrict__ Kout,
                                                    float* __restrict__ Vout) {
    __shared__ unsigned short As[G_MT][T_STR];   // [m][k]
    __shared__ unsigned short Bs[G_NT][T_STR];   // [n][k]  (transposed tile)
    const int tid  = threadIdx.x;
    const int wave = tid >> 5, lane = tid & 31;
    const int half = lane >> 4, l16 = lane & 15;
    const int m0   = blockIdx.x * G_MT;
    const int nb0  = blockIdx.y * G_NT;
    const int wm   = (wave & 3) << 5;   // wave's 32-row group within block tile
    const int wn   = (wave >> 2) << 6;  // wave's 64-col group within block tile

    F32x8 acc[2][4];
#pragma unroll
    for (int mi = 0; mi < 2; mi++)
#pragma unroll
        for (int j = 0; j < 4; j++)
#pragma unroll
            for (int e = 0; e < 8; e++) acc[mi][j].f[e] = 0.f;

    for (int kt = 0; kt < CFEAT; kt += G_KT) {
        __syncthreads();
#pragma unroll
        for (int c2 = 0; c2 < 2; c2++) {         // A stage: 128 rows x 4 chunks(16B)
            int ch = tid * 2 + c2;
            int arow = ch >> 2, aseg = ch & 3;
            *(uint4*)&As[arow][aseg * 8] =
                *(const uint4*)(tok + (size_t)(m0 + arow) * CFEAT + kt + aseg * 8);
        }
#pragma unroll
        for (int c2 = 0; c2 < 2; c2++) {         // B stage: 128 cols x 4 chunks(16B)
            int ch = tid * 2 + c2;
            int bcol = ch >> 2, bseg = ch & 3;
            *(uint4*)&Bs[bcol][bseg * 8] =
                *(const uint4*)(wkvT + (size_t)(nb0 + bcol) * CFEAT + kt + bseg * 8);
        }
        if (kt + G_KT < CFEAT) {                 // gfx1250 global_prefetch of next A tile
            int ch = tid * 2;
            int arow = ch >> 2, aseg = ch & 3;
            __builtin_prefetch(tok + (size_t)(m0 + arow) * CFEAT + kt + G_KT + aseg * 8, 0, 1);
        }
        __syncthreads();

        // A frags: lane needs row (wm + mi*16 + l16), K runs [half*8,+8) and [16+half*8,+8)
        BF16Frag af[2];
#pragma unroll
        for (int mi = 0; mi < 2; mi++) {
            af[mi].q[0] = *(const uint4*)&As[wm + mi * 16 + l16][half * 8];
            af[mi].q[1] = *(const uint4*)&As[wm + mi * 16 + l16][16 + half * 8];
        }
#pragma unroll
        for (int j = 0; j < 4; j++) {
            // B frag: lane needs col (wn+j*16+l16), K runs [half*8,+8) and [16+half*8,+8)
            BF16Frag bf;
            int ncol = wn + j * 16 + l16;
            bf.q[0] = *(const uint4*)&Bs[ncol][half * 8];
            bf.q[1] = *(const uint4*)&Bs[ncol][16 + half * 8];
#pragma unroll
            for (int mi = 0; mi < 2; mi++)
                acc[mi][j].v = __builtin_amdgcn_wmma_f32_16x16x32_bf16(
                    false, af[mi].v, false, bf.v, (short)0, acc[mi][j].v, false, false);
        }
    }

    // epilogue: C/D layout: VGPR r -> (M = r + 8*half, N = l16)
#pragma unroll
    for (int j = 0; j < 4; j++) {
        int gcol = nb0 + wn + j * 16 + l16;
        float bias = (gcol < 256) ? bk[gcol] : bv[gcol - 256];
#pragma unroll
        for (int mi = 0; mi < 2; mi++) {
#pragma unroll
            for (int r = 0; r < 8; r++) {
                int grow = m0 + wm + mi * 16 + (half << 3) + r;
                float val = acc[mi][j].f[r] + bias;
                if (gcol < 256) Kout[(size_t)grow * 256 + gcol] = val;
                else            Vout[(size_t)grow * 256 + (gcol - 256)] = val;
            }
        }
    }
}

// ---------------- 4) joint embedding + Q projection ----------------
__global__ __launch_bounds__(256) void pose_joint(const float* __restrict__ joints3d,
                                                  const float* __restrict__ pred_cam,
                                                  const float* __restrict__ focal,
                                                  const float* __restrict__ w1, const float* __restrict__ b1,
                                                  const float* __restrict__ w2, const float* __restrict__ b2,
                                                  const float* __restrict__ embed,
                                                  const float* __restrict__ wq, const float* __restrict__ bq,
                                                  float* __restrict__ jf_out,
                                                  float* __restrict__ Qout,
                                                  float* __restrict__ j2dn_out) {
    int b = blockIdx.x, c = threadIdx.x;
    __shared__ float ji[NQ][5];
    __shared__ float h1[NQ][BD];
    __shared__ float jf2[NQ][BD];
    if (c < NQ) {
        float sc = pred_cam[b * 3 + 0], tx = pred_cam[b * 3 + 1], ty = pred_cam[b * 3 + 2];
        float fx = focal[b * 2 + 0],    fy = focal[b * 2 + 1];
        float tz = 2.f * fx / (256.f * sc + 1e-9f);
        const float* j3 = joints3d + (size_t)(b * NQ + c) * 3;
        float j0 = j3[0], j1 = j3[1], j2 = j3[2];
        float zc = fmaxf(j2 + tz, 1e-6f);
        float xn = (j0 + tx) / zc, yn = (j1 + ty) / zc;
        float jnx = (xn * fx + 128.f) * (1.f / 128.f) - 1.f;
        float jny = (yn * fy + 128.f) * (1.f / 128.f) - 1.f;
        ji[c][0] = j0; ji[c][1] = j1; ji[c][2] = j2; ji[c][3] = jnx; ji[c][4] = jny;
        j2dn_out[(b * NQ + c) * 2 + 0] = jnx;
        j2dn_out[(b * NQ + c) * 2 + 1] = jny;
    }
    __syncthreads();
    for (int q = 0; q < NQ; q++) {
        float s = b1[c];
#pragma unroll
        for (int i = 0; i < 5; i++) s += ji[q][i] * w1[i * BD + c];
        h1[q][c] = fmaxf(s, 0.f);
    }
    __syncthreads();
    for (int q = 0; q < NQ; q++) {
        float s = b2[c] + embed[q * BD + c];
        for (int i = 0; i < BD; i++) s += h1[q][i] * w2[i * BD + c];
        jf2[q][c] = s;
        jf_out[(size_t)(b * NQ + q) * BD + c] = s;
    }
    __syncthreads();
    for (int q = 0; q < NQ; q++) {
        float s = bq[c];
        for (int i = 0; i < BD; i++) s += jf2[q][i] * wq[i * BD + c];
        Qout[(size_t)(b * NQ + q) * BD + c] = s;
    }
}

// ---------------- 5) attention with geometric bias + softmax ----------------
__global__ __launch_bounds__(256) void pose_attn(const float* __restrict__ Qm,
                                                 const float* __restrict__ Km,
                                                 const float* __restrict__ Vm,
                                                 const float* __restrict__ j2dn,
                                                 float* __restrict__ attnout) {
    int b = blockIdx.x >> 2, h = blockIdx.x & 3;
    __shared__ float Qs[NQ][HDIM];
    __shared__ float attnw[NTOK];
    __shared__ float red[NTOK];
    __shared__ float po[4][HDIM];
    int tid = threadIdx.x;
    for (int idx = tid; idx < NQ * HDIM; idx += 256) {
        int q = idx >> 6, d = idx & 63;
        Qs[q][d] = Qm[(size_t)(b * NQ + q) * BD + h * HDIM + d];
    }
    __syncthreads();
    int n = tid;
    int ix = n & 15, iy = n >> 4;
    float txx = (ix + 0.5f) * 0.125f - 1.f;
    float tyy = (iy + 0.5f) * 0.125f - 1.f;
    const float* Krow  = Km + (size_t)(b * NTOK + n) * BD + h * HDIM;
    int d = tid & 63, g2 = tid >> 6;
    const float* Vbase = Vm + (size_t)(b * NTOK + g2 * 64) * BD + h * HDIM + d;
    for (int q = 0; q < NQ; q++) {
        float s = 0.f;
#pragma unroll 8
        for (int k = 0; k < HDIM; k++) s += Qs[q][k] * Krow[k];
        float jx = j2dn[(b * NQ + q) * 2 + 0], jy = j2dn[(b * NQ + q) * 2 + 1];
        float dx = jx - txx, dy = jy - tyy;
        float logit = s * 0.125f - (dx * dx + dy * dy) * (1.f / 0.72f);
        red[tid] = logit; __syncthreads();
        for (int off = 128; off; off >>= 1) {
            if (tid < off) red[tid] = fmaxf(red[tid], red[tid + off]);
            __syncthreads();
        }
        float mx = red[0]; __syncthreads();
        float e = __expf(logit - mx);
        red[tid] = e; __syncthreads();
        for (int off = 128; off; off >>= 1) {
            if (tid < off) red[tid] += red[tid + off];
            __syncthreads();
        }
        attnw[tid] = e / red[0];
        __syncthreads();
        float s2 = 0.f;
#pragma unroll 8
        for (int i = 0; i < 64; i++) s2 += attnw[g2 * 64 + i] * Vbase[(size_t)i * BD];
        po[g2][d] = s2; __syncthreads();
        if (g2 == 0)
            attnout[(size_t)(b * NQ + q) * BD + h * HDIM + d] =
                po[0][d] + po[1][d] + po[2][d] + po[3][d];
        __syncthreads();
    }
}

// ---------------- 6) out projection + residual + LayerNorm ----------------
__global__ __launch_bounds__(256) void pose_oproj(const float* __restrict__ attnout,
                                                  const float* __restrict__ wo, const float* __restrict__ bo,
                                                  const float* __restrict__ jf,
                                                  const float* __restrict__ ngm, const float* __restrict__ nbm,
                                                  float* __restrict__ outln) {
    int b = blockIdx.x, c = threadIdx.x;
    __shared__ float a[BD];
    __shared__ float red[BD], red2[BD];
    for (int q = 0; q < NQ; q++) {
        a[c] = attnout[(size_t)(b * NQ + q) * BD + c];
        __syncthreads();
        float s = bo[c] + jf[(size_t)(b * NQ + q) * BD + c];
        for (int i = 0; i < BD; i++) s += a[i] * wo[i * BD + c];
        red[c] = s; red2[c] = s * s; __syncthreads();
        for (int off = 128; off; off >>= 1) {
            if (c < off) { red[c] += red[c + off]; red2[c] += red2[c + off]; }
            __syncthreads();
        }
        float m = red[0] * (1.f / BD);
        float v = red2[0] * (1.f / BD) - m * m;
        outln[(size_t)(b * NQ + q) * BD + c] = (s - m) * rsqrtf(v + 1e-5f) * ngm[c] + nbm[c];
        __syncthreads();
    }
}

// ---------------- 7) decoders + Rodrigues + outputs ----------------
__global__ __launch_bounds__(256) void pose_dec(const float* __restrict__ outln,
                                                const float* __restrict__ Wp, const float* __restrict__ bp,
                                                const float* __restrict__ Ws, const float* __restrict__ bs,
                                                const float* __restrict__ Wc, const float* __restrict__ bc,
                                                const float* __restrict__ pose_rotmat,
                                                const float* __restrict__ betas,
                                                const float* __restrict__ cam,
                                                const float* __restrict__ p_ls,
                                                const float* __restrict__ b_ls,
                                                const float* __restrict__ c_ls,
                                                float* __restrict__ out) {
    int b = blockIdx.x, t = threadIdx.x;
    __shared__ float ol[NQ * BD];
    __shared__ float dp[3 * NJ];
    __shared__ float mv[BD];
    for (int i = t; i < NQ * BD; i += 256) ol[i] = outln[(size_t)b * NQ * BD + i];
    __syncthreads();
    if (t < 3 * NJ) {
        float s = bp[t];
        for (int i = 0; i < NQ * BD; i++) s += ol[i] * Wp[(size_t)i * (3 * NJ) + t];
        dp[t] = s * __expf(p_ls[0]);
    }
    float m = 0.f;
    for (int q = 0; q < NQ; q++) m += ol[q * BD + t];
    mv[t] = m * (1.f / NQ);
    __syncthreads();
    if (t < 10) {
        float s = bs[t];
        for (int i = 0; i < BD; i++) s += mv[i] * Ws[i * 10 + t];
        out[BATCH * NJ * 9 + b * 10 + t] = betas[b * 10 + t] + s * __expf(b_ls[0]);
    }
    if (t < 3) {
        float s = bc[t];
        for (int i = 0; i < BD; i++) s += ol[i] * Wc[i * 3 + t];
        out[BATCH * NJ * 9 + BATCH * 10 + b * 3 + t] = cam[b * 3 + t] + s * __expf(c_ls[0]);
    }
    if (t < NJ) {
        float ax = dp[t * 3], ay = dp[t * 3 + 1], az = dp[t * 3 + 2];
        float ang = sqrtf(ax * ax + ay * ay + az * az + 1e-16f);
        float ia = 1.f / ang;
        ax *= ia; ay *= ia; az *= ia;
        float sn = sinf(ang), cs = cosf(ang), oc = 1.f - cs;
        float R[9];
        R[0] = 1.f + oc * (-(ay * ay + az * az));
        R[1] = sn * (-az) + oc * (ax * ay);
        R[2] = sn * ( ay) + oc * (ax * az);
        R[3] = sn * ( az) + oc * (ax * ay);
        R[4] = 1.f + oc * (-(ax * ax + az * az));
        R[5] = sn * (-ax) + oc * (ay * az);
        R[6] = sn * (-ay) + oc * (ax * az);
        R[7] = sn * ( ax) + oc * (ay * az);
        R[8] = 1.f + oc * (-(ax * ax + ay * ay));
        const float* P = pose_rotmat + (size_t)(b * NJ + t) * 9;
        float* O = out + (size_t)(b * NJ + t) * 9;
#pragma unroll
        for (int r = 0; r < 3; r++)
#pragma unroll
            for (int cl = 0; cl < 3; cl++)
                O[r * 3 + cl] = R[r * 3 + 0] * P[0 * 3 + cl]
                              + R[r * 3 + 1] * P[1 * 3 + cl]
                              + R[r * 3 + 2] * P[2 * 3 + cl];
    }
}

// ---------------- workspace layout (bytes) ----------------
#define OFF_TOK   ((size_t)0)                         // 32768*1280*2  = 83,886,080
#define OFF_WKV   ((size_t)83886080)                  // 512*1280*2    =  1,310,720
#define OFF_K     ((size_t)85196800)                  // 32768*256*4   = 33,554,432
#define OFF_V     ((size_t)118751232)                 // 32768*256*4
#define OFF_JF    ((size_t)152305664)                 // 128*21*256*4  =  2,752,512
#define OFF_Q     ((size_t)155058176)
#define OFF_J2DN  ((size_t)157810688)                 // 128*21*2*4    = 21,504 (pad)
#define OFF_ATT   ((size_t)157832192)
#define OFF_OLN   ((size_t)160584704)

extern "C" void kernel_launch(void* const* d_in, const int* in_sizes, int n_in,
                              void* d_out, int out_size, void* d_ws, size_t ws_size,
                              hipStream_t stream) {
    const float* img         = (const float*)d_in[0];
    const float* joints3d    = (const float*)d_in[1];
    const float* pred_cam    = (const float*)d_in[2];
    const float* focal       = (const float*)d_in[3];
    const float* pose_rotmat = (const float*)d_in[4];
    const float* betas       = (const float*)d_in[5];
    const float* cam         = (const float*)d_in[6];
    const float* ln_g        = (const float*)d_in[7];
    const float* ln_b        = (const float*)d_in[8];
    const float* wk          = (const float*)d_in[9];
    const float* bk          = (const float*)d_in[10];
    const float* wv          = (const float*)d_in[11];
    const float* bv          = (const float*)d_in[12];
    const float* je_w1       = (const float*)d_in[13];
    const float* je_b1       = (const float*)d_in[14];
    const float* je_w2       = (const float*)d_in[15];
    const float* je_b2       = (const float*)d_in[16];
    const float* jid_embed   = (const float*)d_in[17];
    const float* wq          = (const float*)d_in[18];
    const float* bq          = (const float*)d_in[19];
    const float* wo          = (const float*)d_in[20];
    const float* bo          = (const float*)d_in[21];
    const float* norm_g      = (const float*)d_in[22];
    const float* norm_b      = (const float*)d_in[23];
    const float* dec_pose_w  = (const float*)d_in[24];
    const float* dec_pose_b  = (const float*)d_in[25];
    const float* dec_shape_w = (const float*)d_in[26];
    const float* dec_shape_b = (const float*)d_in[27];
    const float* dec_cam_w   = (const float*)d_in[28];
    const float* dec_cam_b   = (const float*)d_in[29];
    const float* p_ls        = (const float*)d_in[30];
    const float* b_ls        = (const float*)d_in[31];
    const float* c_ls        = (const float*)d_in[32];

    char* ws = (char*)d_ws;
    unsigned short* tok  = (unsigned short*)(ws + OFF_TOK);
    unsigned short* wkvT = (unsigned short*)(ws + OFF_WKV);
    float* Kout    = (float*)(ws + OFF_K);
    float* Vout    = (float*)(ws + OFF_V);
    float* jf      = (float*)(ws + OFF_JF);
    float* Qm      = (float*)(ws + OFF_Q);
    float* j2dn    = (float*)(ws + OFF_J2DN);
    float* attnout = (float*)(ws + OFF_ATT);
    float* outln   = (float*)(ws + OFF_OLN);

    pose_pack_wkv<<<512, 256, 0, stream>>>(wk, wv, wkvT);
    pose_ln<<<BATCH * (NTOK / 64), 256, 0, stream>>>(img, ln_g, ln_b, tok);
    pose_joint<<<BATCH, 256, 0, stream>>>(joints3d, pred_cam, focal,
                                          je_w1, je_b1, je_w2, je_b2, jid_embed,
                                          wq, bq, jf, Qm, j2dn);
    dim3 gemm_grid(MTOT / G_MT, 512 / G_NT);
    pose_kv_gemm<<<gemm_grid, 256, 0, stream>>>(tok, wkvT, bk, bv, Kout, Vout);
    pose_attn<<<BATCH * NHEAD, 256, 0, stream>>>(Qm, Kout, Vout, j2dn, attnout);
    pose_oproj<<<BATCH, 256, 0, stream>>>(attnout, wo, bo, jf, norm_g, norm_b, outln);
    pose_dec<<<BATCH, 256, 0, stream>>>(outln, dec_pose_w, dec_pose_b,
                                        dec_shape_w, dec_shape_b, dec_cam_w, dec_cam_b,
                                        pose_rotmat, betas, cam, p_ls, b_ls, c_ls,
                                        (float*)d_out);
}
